// SaConvLSTM2DCell_28441273434238
// MI455X (gfx1250) — matrix operand order repaired and checked by reading, not compiled, not run on hardware
//
#include <hip/hip_runtime.h>
#include <hip/hip_bf16.h>
#include <math.h>
#include <stdint.h>

// ---- problem dims (fixed by reference) ----
#define BB   16
#define HH   128
#define WW   128
#define CC   64      // Cin
#define FF   64      // F
#define NG   256     // 4F gate channels
#define KC   128     // combined input channels (Cin + F)
#define NPIX (BB*HH*WW)          // 262144
#define NPF  (NPIX*FF)           // 16777216 elements per output tensor

// K2 LDS layout (bytes); rows padded to 136 bf16 (272B stride -> conflict-free b128)
#define KPAD    136
#define SMA_SZ  (3 * 130 * KPAD * 2)          // 106,080 B  (act halo: 3 rows x 130 px)
#define SMB_SZ  (NG * KPAD * 2)               //  69,632 B  (one tap's weights, padded)
#define SMEM_SZ (SMA_SZ + 2 * SMB_SZ)         // 245,344 B  (smZ 128*256*4 = 131,072 aliases)

typedef __attribute__((ext_vector_type(16))) __bf16 bf16x16;
typedef __attribute__((ext_vector_type(8)))  __bf16 bf16x8;
typedef __attribute__((ext_vector_type(8)))  float  f32x8;

union Frag16 { bf16x16 v; bf16x8 h[2]; };

__device__ __forceinline__ float hsig(float x) {
    return fminf(fmaxf(0.2f * x + 0.5f, 0.0f), 1.0f);
}
__device__ __forceinline__ float sigm(float x) {
    return 1.0f / (1.0f + __expf(-x));
}

// gfx1250 async global->LDS copy (no VGPR data path; tracked by ASYNCcnt)
__device__ __forceinline__ void async_b128_to_lds(void* lds_dst, const void* gsrc) {
    uint32_t lds_off = (uint32_t)(uintptr_t)lds_dst;   // low 32 bits = LDS address
    uint64_t gaddr   = (uint64_t)(uintptr_t)gsrc;
    asm volatile("global_load_async_to_lds_b128 %0, %1, off"
                 :: "v"(lds_off), "v"(gaddr)
                 : "memory");
}
__device__ __forceinline__ void wait_async0() {
    asm volatile("s_wait_asynccnt 0x0" ::: "memory");
}

// ------------------------------------------------------------------
// K0: pack weights -> bf16 [tap(9)][n(256)][k(128)]  (k: 0..63 = Cin, 64..127 = recurrent F)
// ------------------------------------------------------------------
__global__ void k_pack_wgt(const float* __restrict__ kern,      // [3][3][64][256]
                           const float* __restrict__ rkern,     // [3][3][64][256]
                           __bf16* __restrict__ wgt) {
    int idx = blockIdx.x * 256 + threadIdx.x;       // (t*256+n)*128+k
    if (idx >= 9 * NG * KC) return;
    int k = idx & 127;
    int n = (idx >> 7) & 255;
    int t = idx >> 15;
    float v = (k < CC) ? kern[(t * CC + k) * NG + n]
                       : rkern[(t * CC + (k - CC)) * NG + n];
    wgt[idx] = (__bf16)v;
}

// ------------------------------------------------------------------
// K1: pack activations -> bf16 [B,H,W,128] = concat(inputs, h0) on channels
// ------------------------------------------------------------------
__global__ void k_pack_act(const float* __restrict__ x,   // [B,H,W,64]
                           const float* __restrict__ h0,  // [B,H,W,64]
                           __bf16* __restrict__ act) {
    size_t idx = (size_t)blockIdx.x * 256 + threadIdx.x;
    if (idx >= (size_t)NPIX * KC) return;
    size_t p = idx >> 7;
    int    c = (int)(idx & 127);
    float  v = (c < CC) ? x[p * CC + c] : h0[p * FF + (c - CC)];
    act[idx] = (__bf16)v;
}

// ------------------------------------------------------------------
// K2: fused implicit-GEMM ConvLSTM gate kernel (WMMA bf16) + gate epilogue.
// Block: 512 threads = 16 waves. Tile: one full row (128 px) x 256 gate ch.
// Wave w: M-tile mt = w&7 (16 px), N-half = w>>3 (128 ch = 8 N tiles).
// Per-tap weights double-buffered through LDS via async global->LDS copies.
// Per k-block: batch-load A + all 8 B fragments, then 8 wmmas -> counted dscnt waits.
// ------------------------------------------------------------------
__global__ __launch_bounds__(512)
void k_gates_wmma(const __bf16* __restrict__ act,   // [B,H,W,128] bf16
                  const __bf16* __restrict__ wgt,   // [9][256][128] bf16
                  const float*  __restrict__ bias,  // [256]
                  const float*  __restrict__ c0,    // [B,H,W,64]
                  float* __restrict__ out_c,        // [B,H,W,64]
                  float* __restrict__ out_h) {      // [B,H,W,64]
    extern __shared__ __align__(16) char smem[];
    __bf16* smA  = (__bf16*)smem;                       // [3][130][KPAD]
    __bf16* smB0 = (__bf16*)(smem + SMA_SZ);            // [256][KPAD]
    __bf16* smB1 = (__bf16*)(smem + SMA_SZ + SMB_SZ);   // [256][KPAD]
    float (*smZ)[NG] = (float (*)[NG])smem;             // [128][256] (aliases, post-GEMM)

    const int tid  = threadIdx.x;
    const int lane = tid & 31;
    const int wid  = tid >> 5;
    const int lm   = lane & 15;
    const int lh   = lane >> 4;
    const int mt    = wid & 7;
    const int nhalf = wid >> 3;

    const int bb = blockIdx.x >> 7;           // batch
    const int hy = blockIdx.x & 127;          // output row

    // ---- async-stage tap 0 weights into smB0 (no VGPR data path) ----
    for (int idx = tid; idx < NG * 16; idx += 512) {
        int n = idx >> 4, cch = idx & 15;
        async_b128_to_lds(smB0 + (size_t)n * KPAD + cch * 8,
                          wgt + (size_t)n * KC + cch * 8);
    }
    // ---- stage 3x130x128 bf16 act halo into LDS (zero-padded, padded stride) ----
    for (int idx = tid; idx < 3 * 130 * 16; idx += 512) {
        int cch = idx & 15;
        int wi  = (idx >> 4) % 130;
        int r   = idx / (130 * 16);
        int sy  = hy + r - 1;
        int sx  = wi - 1;
        int4 val = make_int4(0, 0, 0, 0);
        if (sy >= 0 && sy < HH && sx >= 0 && sx < WW) {
            size_t off = ((((size_t)bb * HH + sy) * WW + sx) * KC) + (size_t)cch * 8;
            val = *(const int4*)(act + off);
        }
        *(int4*)(smA + ((size_t)r * 130 + wi) * KPAD + cch * 8) = val;
    }
    wait_async0();
    __syncthreads();

    f32x8 acc[8];
#pragma unroll
    for (int i = 0; i < 8; ++i) acc[i] = (f32x8){0.f,0.f,0.f,0.f,0.f,0.f,0.f,0.f};

    for (int t = 0; t < 9; ++t) {
        __bf16* cur = (t & 1) ? smB1 : smB0;
        __bf16* nxt = (t & 1) ? smB0 : smB1;

        // kick off next tap's weight copy straight into the other LDS buffer
        if (t < 8) {
            const __bf16* wsrc = wgt + (size_t)(t + 1) * NG * KC;
#pragma unroll
            for (int j = 0; j < 8; ++j) {
                int idx = tid + j * 512;
                async_b128_to_lds(nxt + (size_t)(idx >> 4) * KPAD + (idx & 15) * 8,
                                  wsrc + (size_t)(idx >> 4) * KC + (idx & 15) * 8);
            }
        }

        const int dy = t / 3, dx = t % 3;
#pragma unroll
        for (int cb = 0; cb < 4; ++cb) {
            // A fragment: 16-bit A 16x32 layout (lane = M, K-half by lane>>4)
            Frag16 af;
            const __bf16* ap = smA + ((size_t)dy * 130 + mt * 16 + lm + dx) * KPAD
                                   + cb * 32 + lh * 8;
            af.h[0] = *(const bf16x8*)ap;         // K = half*8 .. +7
            af.h[1] = *(const bf16x8*)(ap + 16);  // K = 16 + half*8 .. +7

            // Batch-load all 8 B fragments (16-bit B 32x16 layout), then issue
            // the wmmas: waitcnt pass can count down DScnt instead of draining.
            const __bf16* bbase = cur + (size_t)(nhalf * 128 + lm) * KPAD + cb * 32 + lh * 8;
            Frag16 bf[8];
#pragma unroll
            for (int nt = 0; nt < 8; ++nt) {
                const __bf16* bp = bbase + (size_t)(nt * 16) * KPAD;
                bf[nt].h[0] = *(const bf16x8*)bp;
                bf[nt].h[1] = *(const bf16x8*)(bp + 16);
            }
#pragma unroll
            for (int nt = 0; nt < 8; ++nt) {
                acc[nt] = __builtin_amdgcn_wmma_f32_16x16x32_bf16(
                    false, af.v, false, bf[nt].v, (short)0, acc[nt], false, false);
            }
        }

        if (t < 8) wait_async0();   // next tap's LDS writes done before the barrier
        __syncthreads();
    }

    // ---- spill C tiles to LDS (aliases stage buffers; GEMM reads done) ----
#pragma unroll
    for (int nt = 0; nt < 8; ++nt) {
#pragma unroll
        for (int r = 0; r < 8; ++r) {
            smZ[mt * 16 + lh * 8 + r][nhalf * 128 + nt * 16 + lm] = acc[nt][r];
        }
    }
    __syncthreads();

    // ---- fused LSTM gate epilogue: 128 px x 64 F channels ----
    for (int idx = tid; idx < 128 * FF; idx += 512) {
        int pl = idx >> 6;
        int f  = idx & 63;
        size_t g = (((size_t)bb * HH + hy) * WW + pl);
        float zi_ = smZ[pl][f]        + bias[f];
        float zf_ = smZ[pl][64 + f]   + bias[64 + f];
        float zc_ = smZ[pl][128 + f]  + bias[128 + f];
        float zo_ = smZ[pl][192 + f]  + bias[192 + f];
        float ig = hsig(zi_);
        float fg = hsig(zf_);
        float og = hsig(zo_);
        float cv = fg * c0[g * FF + f] + ig * tanhf(zc_);
        float hv = og * tanhf(cv);
        out_c[g * FF + f] = cv;
        out_h[g * FF + f] = hv;
    }
}

// ------------------------------------------------------------------
// K3: per-pixel channel contractions: v_h,k_h,q_h (from h), k_m,v_m (from m0)
// ------------------------------------------------------------------
__global__ void k_sam_proj(const float* __restrict__ h,
                           const float* __restrict__ m0,
                           const float* __restrict__ saw,   // [64][5]
                           float* __restrict__ v_h, float* __restrict__ k_h,
                           float* __restrict__ q_h, float* __restrict__ k_m,
                           float* __restrict__ v_m) {
    size_t p = (size_t)blockIdx.x * 256 + threadIdx.x;
    if (p >= NPIX) return;
    float a0 = 0.f, a1 = 0.f, a2 = 0.f, a3 = 0.f, a4 = 0.f;
    for (int c = 0; c < FF; ++c) {
        float hv = h[p * FF + c];
        float mv = m0[p * FF + c];
        const float* w = saw + c * 5;
        a0 += hv * w[0]; a1 += hv * w[1]; a2 += hv * w[2];
        a3 += mv * w[3]; a4 += mv * w[4];
    }
    v_h[p] = a0; k_h[p] = a1; q_h[p] = a2; k_m[p] = a3; v_m[p] = a4;
}

// ------------------------------------------------------------------
// K4: attention rows. One block per (b, i): scores -> softmax -> weighted sum.
// ------------------------------------------------------------------
__global__ __launch_bounds__(128)
void k_sam_attn(const float* __restrict__ q_h, const float* __restrict__ k_h,
                const float* __restrict__ k_m, const float* __restrict__ v_h,
                const float* __restrict__ v_m,
                float* __restrict__ z_h, float* __restrict__ z_m) {
    __shared__ float sQ[128], sM[128], sH[128], sR[128];
    const int t = threadIdx.x;
    const int b = blockIdx.x >> 7;
    const int i = blockIdx.x & 127;
    const size_t base = (size_t)b * HH * WW;

    sQ[t] = q_h[base + (size_t)i * WW + t];
    __syncthreads();

    float sm = 0.f, sh = 0.f;
    for (int w = 0; w < WW; ++w) {
        float q = sQ[w];
        sm += q * k_m[base + (size_t)w * WW + t];
        sh += q * k_h[base + (size_t)w * WW + t];
    }
    sM[t] = sm; sH[t] = sh;

    // softmax over j for both score rows
    sR[t] = sm; __syncthreads();
    for (int s = 64; s > 0; s >>= 1) { if (t < s) sR[t] = fmaxf(sR[t], sR[t + s]); __syncthreads(); }
    float mxM = sR[0]; __syncthreads();
    sR[t] = sh; __syncthreads();
    for (int s = 64; s > 0; s >>= 1) { if (t < s) sR[t] = fmaxf(sR[t], sR[t + s]); __syncthreads(); }
    float mxH = sR[0]; __syncthreads();

    float em = __expf(sm - mxM);
    float eh = __expf(sh - mxH);
    sM[t] = em; sH[t] = eh;
    sR[t] = em; __syncthreads();
    for (int s = 64; s > 0; s >>= 1) { if (t < s) sR[t] += sR[t + s]; __syncthreads(); }
    float invM = 1.0f / sR[0]; __syncthreads();
    sR[t] = eh; __syncthreads();
    for (int s = 64; s > 0; s >>= 1) { if (t < s) sR[t] += sR[t + s]; __syncthreads(); }
    float invH = 1.0f / sR[0]; __syncthreads();

    float zm = 0.f, zh = 0.f;
    for (int j = 0; j < WW; ++j) {
        zm += sM[j] * v_m[base + (size_t)j * WW + t];
        zh += sH[j] * v_h[base + (size_t)j * WW + t];
    }
    z_m[base + (size_t)i * WW + t] = zm * invM;
    z_h[base + (size_t)i * WW + t] = zh * invH;
}

// ------------------------------------------------------------------
// K5: zi = conv3x3(stack([z_h, z_m], -1), kernel_z[3,3,2,64]), SAME
// ------------------------------------------------------------------
__global__ void k_zi_conv(const float* __restrict__ z_h, const float* __restrict__ z_m,
                          const float* __restrict__ kz,  // [3][3][2][64]
                          float* __restrict__ zi) {
    size_t idx = (size_t)blockIdx.x * 256 + threadIdx.x;
    if (idx >= (size_t)NPIX * FF) return;
    size_t p  = idx >> 6;
    int    oc = (int)(idx & 63);
    int x = (int)(p & 127);
    int y = (int)((p >> 7) & 127);
    int b = (int)(p >> 14);
    float acc = 0.f;
    for (int ky = 0; ky < 3; ++ky) {
        int yy = y + ky - 1;
        if (yy < 0 || yy >= HH) continue;
        for (int kx = 0; kx < 3; ++kx) {
            int xx = x + kx - 1;
            if (xx < 0 || xx >= WW) continue;
            size_t n = ((size_t)b * HH + yy) * WW + xx;
            int tk = ky * 3 + kx;
            acc += z_h[n] * kz[(tk * 2 + 0) * FF + oc];
            acc += z_m[n] * kz[(tk * 2 + 1) * FF + oc];
        }
    }
    zi[p * FF + oc] = acc;
}

// ------------------------------------------------------------------
// K6: six depthwise 3x3 convs + final elementwise -> h_hat, m_t
// ------------------------------------------------------------------
__global__ void k_dw_final(const float* __restrict__ zi, const float* __restrict__ h,
                           const float* __restrict__ m0,
                           const float* __restrict__ dw,   // [3][3][64][6]
                           const float* __restrict__ sab,  // [192]
                           float* __restrict__ out_hhat, float* __restrict__ out_mt) {
    size_t idx = (size_t)blockIdx.x * 256 + threadIdx.x;
    if (idx >= (size_t)NPIX * FF) return;
    size_t p = idx >> 6;
    int    f = (int)(idx & 63);
    int x = (int)(p & 127);
    int y = (int)((p >> 7) & 127);
    int b = (int)(p >> 14);
    float s0 = 0.f, s1 = 0.f, s2 = 0.f, s3 = 0.f, s4 = 0.f, s5 = 0.f;
    for (int ky = 0; ky < 3; ++ky) {
        int yy = y + ky - 1;
        if (yy < 0 || yy >= HH) continue;
        for (int kx = 0; kx < 3; ++kx) {
            int xx = x + kx - 1;
            if (xx < 0 || xx >= WW) continue;
            size_t n = ((size_t)b * HH + yy) * WW + xx;
            float zv = zi[n * FF + f];
            float hv = h[n * FF + f];
            const float* k6 = dw + ((size_t)(ky * 3 + kx) * FF + f) * 6;
            s0 += zv * k6[0]; s1 += hv * k6[1];
            s2 += zv * k6[2]; s3 += hv * k6[3];
            s4 += zv * k6[4]; s5 += hv * k6[5];
        }
    }
    float i2 = sigm(s0 + s1 + sab[f]);
    float g  = tanhf(s2 + s3 + sab[64 + f]);
    float o2 = sigm(s4 + s5 + sab[128 + f]);
    float mt = (1.0f - i2) * m0[p * FF + f] + i2 * g;
    out_mt[p * FF + f]   = mt;
    out_hhat[p * FF + f] = mt * o2;
}

// ------------------------------------------------------------------
extern "C" void kernel_launch(void* const* d_in, const int* in_sizes, int n_in,
                              void* d_out, int out_size, void* d_ws, size_t ws_size,
                              hipStream_t stream) {
    const float* inputs = (const float*)d_in[0];
    const float* h0     = (const float*)d_in[1];
    const float* c0     = (const float*)d_in[2];
    const float* m0     = (const float*)d_in[3];
    const float* kern   = (const float*)d_in[4];
    const float* rkern  = (const float*)d_in[5];
    const float* bias   = (const float*)d_in[6];
    const float* sak    = (const float*)d_in[7];   // [1,1,64,5]
    const float* kz     = (const float*)d_in[8];   // [3,3,2,64]
    const float* dwk    = (const float*)d_in[9];   // [3,3,64,6]
    const float* sab    = (const float*)d_in[10];  // [192]

    float* out_hhat = (float*)d_out;
    float* out_c    = (float*)d_out + NPF;
    float* out_mt   = (float*)d_out + 2 * (size_t)NPF;

    // workspace carve-up
    char* ws = (char*)d_ws;
    __bf16* act = (__bf16*)ws;                         ws += (size_t)NPIX * KC * 2;   // 64 MB
    __bf16* wgt = (__bf16*)ws;                         ws += (size_t)9 * NG * KC * 2; // 0.56 MB
    float*  hbuf = (float*)ws;                         ws += (size_t)NPF * 4;         // 64 MB
    float*  v_h  = (float*)ws;                         ws += (size_t)NPIX * 4;
    float*  k_h  = (float*)ws;                         ws += (size_t)NPIX * 4;
    float*  q_h  = (float*)ws;                         ws += (size_t)NPIX * 4;
    float*  k_m  = (float*)ws;                         ws += (size_t)NPIX * 4;
    float*  v_m  = (float*)ws;                         ws += (size_t)NPIX * 4;
    float*  z_h  = (float*)ws;                         ws += (size_t)NPIX * 4;
    float*  z_m  = (float*)ws;                         ws += (size_t)NPIX * 4;
    float*  zi   = (float*)ws;                         ws += (size_t)NPF * 4;         // 64 MB

    // K0/K1: pack operands to bf16
    k_pack_wgt<<<(9 * NG * KC + 255) / 256, 256, 0, stream>>>(kern, rkern, wgt);
    k_pack_act<<<(int)(((size_t)NPIX * KC + 255) / 256), 256, 0, stream>>>(inputs, h0, act);

    // K2: fused implicit-GEMM ConvLSTM gates (WMMA, async-LDS-staged weights) -> c, h
    k_gates_wmma<<<BB * HH, 512, SMEM_SZ, stream>>>(act, wgt, bias, c0, out_c, hbuf);

    // K3: SAM projections
    k_sam_proj<<<(NPIX + 255) / 256, 256, 0, stream>>>(hbuf, m0, sak, v_h, k_h, q_h, k_m, v_m);

    // K4: attention (softmax rows)
    k_sam_attn<<<BB * HH, 128, 0, stream>>>(q_h, k_h, k_m, v_h, v_m, z_h, z_m);

    // K5: zi conv (2 -> 64 channels)
    k_zi_conv<<<(int)(((size_t)NPIX * FF + 255) / 256), 256, 0, stream>>>(z_h, z_m, kz, zi);

    // K6: depthwise convs + final gate math -> h_hat, m_t
    k_dw_final<<<(int)(((size_t)NPIX * FF + 255) / 256), 256, 0, stream>>>(zi, hbuf, m0, dwk, sab,
                                                                           out_hhat, out_mt);
}
